// Attention_43963285242717
// MI455X (gfx1250) — matrix operand order repaired and checked by reading, not compiled, not run on hardware
//
#include <hip/hip_runtime.h>
#include <hip/hip_bf16.h>

// ---------------------------------------------------------------------------
// MI455X (gfx1250) attention block: bf16 WMMA pipeline, fp32 accumulate,
// async global->LDS double-buffered staging (ASYNCcnt), ds_load fragments.
//   x[2048,4608] -> Q/K/V proj (WMMA GEMM) -> RoPE -> flash attention (WMMA)
//   -> output projection (WMMA GEMM) -> out[2048,4608] fp32
// ---------------------------------------------------------------------------

typedef __attribute__((ext_vector_type(16))) __bf16 v16bf;
typedef __attribute__((ext_vector_type(8)))  __bf16 v8bf;
typedef __attribute__((ext_vector_type(4)))  __bf16 v4bf;
typedef __attribute__((ext_vector_type(8)))  float  v8f;
typedef __attribute__((ext_vector_type(4)))  float  v4f;

#define L_SEQ   2048
#define DMODEL  4608
#define NHEADS  32
#define NKV     16
#define HD      128
#define QDIM    (NHEADS * HD)   // 4096
#define KVDIM   (NKV * HD)      // 2048

// ---- CDNA5 async global->LDS copy (ASYNCcnt-tracked, no VGPR round trip) --
// ISA 10.x: GLOBAL_LOAD_ASYNC_TO_LDS_B128; VDST = LDS byte address, VADDR =
// 64-bit global address.  Generic pointers to LDS carry the LDS offset in
// their low 32 bits (aperture in the high bits), so a truncating cast gives
// the LDS address the instruction wants.
__device__ __forceinline__ void async_copy16(const __bf16* g, __bf16* l) {
    unsigned lds_addr = (unsigned)(size_t)l;
    asm volatile("global_load_async_to_lds_b128 %0, %1, off"
                 :: "v"(lds_addr), "v"(g) : "memory");
}
__device__ __forceinline__ void wait_async_le4() {
    asm volatile("s_wait_asynccnt 4" ::: "memory");
}
__device__ __forceinline__ void wait_async_le0() {
    asm volatile("s_wait_asynccnt 0" ::: "memory");
}

// ---- WMMA fragment loaders (k = 16h+s order in both A and B, wave32) -------------------
// A-matrix 16x32 bf16: lane r=lane&15 holds row r; kb=(lane>=16)?16:0;
//   VGPR0..3 = K[kb..kb+7], VGPR4..7 = K[kb+8..kb+15]  (two 16B chunks)
__device__ __forceinline__ v16bf load_a_frag(const __bf16* tile, int ld, int lane) {
    const int r  = lane & 15;
    const int kb = (lane >> 4) << 4;          // 0 or 16
    const __bf16* p = tile + (size_t)r * ld + kb;
    union { v16bf v; v8bf h[2]; } u;
    u.h[0] = *(const v8bf*)(p);
    u.h[1] = *(const v8bf*)(p + 8);
    return u.v;
}

// B-matrix 32x16 bf16: lane n=lane&15 holds column n; kb=(lane>=16)?16:0;
//   frag elem e = B[kb+e, n].  With B stored as W[n, k] (k contiguous) this is
//   one contiguous 32B load per lane.
__device__ __forceinline__ v16bf load_b_frag(const __bf16* tile, int ld, int lane) {
    const int n  = lane & 15;
    const int kb = (lane >> 4) << 4;          // 0 or 16
    return *(const v16bf*)(tile + (size_t)n * ld + kb);
}

__device__ __forceinline__ v8f wmma_bf16(v16bf a, v16bf b, v8f c) {
    return __builtin_amdgcn_wmma_f32_16x16x32_bf16(
        /*neg_a=*/false, a, /*neg_b=*/false, b,
        /*c_mod=*/(short)0, c, /*reuse_a=*/false, /*reuse_b=*/false);
}

// ---- fp32 -> bf16 conversion (x4 vectorized) ------------------------------
__global__ void cvt_f32_bf16x4(const float* __restrict__ s, __bf16* __restrict__ d, int n4) {
    int i = blockIdx.x * blockDim.x + threadIdx.x;
    if (i >= n4) return;
    v4f v = *(const v4f*)(s + (size_t)i * 4);
    v4bf b;
#pragma unroll
    for (int j = 0; j < 4; ++j) b[j] = (__bf16)v[j];
    *(v4bf*)(d + (size_t)i * 4) = b;
}

// ---- GEMM: C[M,N] = A[M,K] * W[N,K]^T  (bf16 in, f32 acc) -----------------
// Block = 256 threads = 8 waves in a 4(M) x 2(N) grid; block tile 128x128.
// Wave computes 32(M) x 64(N): 2 A-frags x 4 B-frags = 8 WMMAs per K-step.
// A/B 128x32 tiles double-buffered in LDS via async copies.
// grid = (N/128, M/128).
__global__ __launch_bounds__(256) void gemm_xwT(
    const __bf16* __restrict__ A, const __bf16* __restrict__ W,
    __bf16* __restrict__ Cb, float* __restrict__ Cf,
    int M, int N, int K)
{
    __shared__ __align__(32) __bf16 abuf[2][128 * 32];   // 8 KB each
    __shared__ __align__(32) __bf16 bbuf[2][128 * 32];

    const int tid    = threadIdx.x;
    const int lane   = tid & 31;
    const int wave   = tid >> 5;
    const int wm     = wave >> 1;               // 0..3
    const int wn     = wave & 1;                // 0..1
    const int mblock = blockIdx.y << 7;
    const int nblock = blockIdx.x << 7;

    // 128x32 tile = 512 chunks of 16B; thread stages chunks tid and tid+256.
    // chunk idx -> row = idx/4, col = (idx&3)*8; LDS offset = idx*8 elements.
    auto stage = [&](int k0, int buf) {
#pragma unroll
        for (int s = 0; s < 2; ++s) {
            const int idx = tid + (s << 8);
            const int row = idx >> 2, col = (idx & 3) << 3;
            async_copy16(A + (size_t)(mblock + row) * K + k0 + col, &abuf[buf][idx << 3]);
            async_copy16(W + (size_t)(nblock + row) * K + k0 + col, &bbuf[buf][idx << 3]);
        }
    };

    stage(0, 0);
    v8f acc[2][4] = {};
    const int nk = K >> 5;
    for (int ks = 0; ks < nk; ++ks) {
        const int buf = ks & 1;
        if (ks + 1 < nk) { stage((ks + 1) << 5, buf ^ 1); wait_async_le4(); }
        else             { wait_async_le0(); }
        __syncthreads();                        // all waves' tiles visible

        v16bf aF[2], bF[4];
#pragma unroll
        for (int i = 0; i < 2; ++i)
            aF[i] = load_a_frag(&abuf[buf][(wm * 32 + i * 16) * 32], 32, lane);
#pragma unroll
        for (int j = 0; j < 4; ++j)
            bF[j] = load_b_frag(&bbuf[buf][(wn * 64 + j * 16) * 32], 32, lane);
#pragma unroll
        for (int i = 0; i < 2; ++i)
#pragma unroll
            for (int j = 0; j < 4; ++j)
                acc[i][j] = wmma_bf16(aF[i], bF[j], acc[i][j]);

        __syncthreads();                        // reads done before next async overwrite
    }

    // C/D layout: VGPR e, lanes 0-15 -> row e; lanes 16-31 -> row e+8; col = lane&15
    const int grp = lane >> 4, col = lane & 15;
#pragma unroll
    for (int i = 0; i < 2; ++i)
#pragma unroll
        for (int j = 0; j < 4; ++j) {
            const int n = nblock + wn * 64 + j * 16 + col;
#pragma unroll
            for (int e = 0; e < 8; ++e) {
                const int m = mblock + wm * 32 + i * 16 + e + (grp << 3);
                if (Cb) Cb[(size_t)m * N + n] = (__bf16)acc[i][j][e];
                else    Cf[(size_t)m * N + n] = acc[i][j][e];
            }
        }
}

// ---- RoPE in place on bf16 [L, nheads*128], fused scale -------------------
__global__ void rope_inplace(__bf16* __restrict__ t, int nheads, float scale) {
    int idx = blockIdx.x * blockDim.x + threadIdx.x;  // L * nheads * 64
    int j = idx & 63;
    int h = (idx >> 6) % nheads;
    int l = (idx >> 6) / nheads;
    if (l >= L_SEQ) return;
    const size_t row = (size_t)l * nheads * HD + (size_t)h * HD;
    float inv = __expf(-0.14391156856f * (float)j);   // 10000^(-j/64)
    float ang = (float)l * inv;
    float sn, cs;
    sincosf(ang, &sn, &cs);
    float x1 = (float)t[row + j];
    float x2 = (float)t[row + j + 64];
    t[row + j]      = (__bf16)((x1 * cs - x2 * sn) * scale);
    t[row + j + 64] = (__bf16)((x2 * cs + x1 * sn) * scale);
}

// ---- V transpose: [L, 2048] -> [2048, L] so PV B-frags are contiguous -----
__global__ void transpose_bf16(const __bf16* __restrict__ s, __bf16* __restrict__ d) {
    int idx = blockIdx.x * blockDim.x + threadIdx.x;   // L * KVDIM
    int r = idx / KVDIM, c = idx - r * KVDIM;
    if (r >= L_SEQ) return;
    d[(size_t)c * L_SEQ + r] = s[idx];
}

// ---- Flash attention: block = (head, 128-query chunk), 8 waves x 16 rows --
// K (32x128) and V^T (128x32) tiles are shared by all 8 waves -> staged once
// per key-block into LDS ping-pong buffers via async copies.
__global__ __launch_bounds__(256) void flash_attn(
    const __bf16* __restrict__ Q, const __bf16* __restrict__ K,
    const __bf16* __restrict__ Vt, __bf16* __restrict__ O)
{
    __shared__ __align__(32) __bf16 kbuf[2][32 * HD];   // keys x d, ld=128 (8 KB)
    __shared__ __align__(32) __bf16 vbuf[2][HD * 32];   // d x keys, ld=32  (8 KB)
    __shared__ __align__(32) __bf16 plds[8][16 * 32];   // per-wave P re-layout slab

    const int tid   = threadIdx.x;
    const int lane  = tid & 31;
    const int wave  = tid >> 5;
    const int h     = blockIdx.y;
    const int g     = h >> 1;                  // kv head (REPEATS = 2)
    const int qc    = blockIdx.x;
    const int qbase = (qc << 7) + (wave << 4);
    const int grp = lane >> 4, col = lane & 15;

    const __bf16* Kh  = K  + (size_t)g * HD;              // rows stride KVDIM
    const __bf16* Vth = Vt + (size_t)(g * HD) * L_SEQ;    // rows stride L_SEQ

    // stage one 32-key block: K tile 32x128 and V^T tile 128x32 (16 KB total)
    auto stage = [&](int kbase, int buf) {
#pragma unroll
        for (int s = 0; s < 2; ++s) {
            const int idx = tid + (s << 8);                   // 0..511
            const int kr = idx >> 4, kc = (idx & 15) << 3;    // 32 x (128/8)
            async_copy16(Kh + (size_t)(kbase + kr) * KVDIM + kc, &kbuf[buf][idx << 3]);
            const int vr = idx >> 2, vc = (idx & 3) << 3;     // 128 x (32/8)
            async_copy16(Vth + (size_t)vr * L_SEQ + kbase + vc, &vbuf[buf][idx << 3]);
        }
    };

    // Q tile 16x128 as 4 A-frags (SCALE already folded in by RoPE)
    v16bf qF[4];
#pragma unroll
    for (int c = 0; c < 4; ++c)
        qF[c] = load_a_frag(Q + (size_t)qbase * QDIM + h * HD + c * 32, QDIM, lane);

    v8f acc[8] = {};
    float mrow[8], lrow[8];
#pragma unroll
    for (int e = 0; e < 8; ++e) { mrow[e] = -1e30f; lrow[e] = 0.f; }

    const int nkt = (qc + 1) << 2;             // 32-key blocks to causal frontier
    stage(0, 0);
    for (int kt = 0; kt < nkt; ++kt) {
        const int kbase = kt << 5;
        const int buf   = kt & 1;
        if (kt + 1 < nkt) { stage((kt + 1) << 5, buf ^ 1); wait_async_le4(); }
        else              { wait_async_le0(); }
        __syncthreads();

        // S = Q (16x128) . K^T (128x32): two 16-key subtiles, 4 d-chunks each
        v8f s0 = {}, s1 = {};
#pragma unroll
        for (int c = 0; c < 4; ++c) {
            v16bf kf0 = load_b_frag(&kbuf[buf][ 0 * HD + c * 32], HD, lane);
            v16bf kf1 = load_b_frag(&kbuf[buf][16 * HD + c * 32], HD, lane);
            s0 = wmma_bf16(qF[c], kf0, s0);
            s1 = wmma_bf16(qF[c], kf1, s1);
        }

        // softcap + causal mask + online softmax (row stats via 16-lane xor tree)
        float alpha[8];
#pragma unroll
        for (int e = 0; e < 8; ++e) {
            const int qrow = qbase + e + (grp << 3);
            float v0 = 50.f * tanhf(s0[e] * 0.02f);
            float v1 = 50.f * tanhf(s1[e] * 0.02f);
            if (kbase + col > qrow)      v0 = -1e30f;
            if (kbase + 16 + col > qrow) v1 = -1e30f;
            float mx = fmaxf(v0, v1);
#pragma unroll
            for (int off = 1; off < 16; off <<= 1) mx = fmaxf(mx, __shfl_xor(mx, off, 32));
            const float mnew = fmaxf(mrow[e], mx);
            const float a    = __expf(mrow[e] - mnew);
            mrow[e] = mnew; alpha[e] = a;
            const float p0 = __expf(v0 - mnew);
            const float p1 = __expf(v1 - mnew);
            float rs = p0 + p1;
#pragma unroll
            for (int off = 1; off < 16; off <<= 1) rs += __shfl_xor(rs, off, 32);
            lrow[e] = lrow[e] * a + rs;
            const int prow = e + (grp << 3);
            plds[wave][prow * 32 + col]      = (__bf16)p0;
            plds[wave][prow * 32 + col + 16] = (__bf16)p1;
        }
#pragma unroll
        for (int dt = 0; dt < 8; ++dt)
#pragma unroll
            for (int e = 0; e < 8; ++e) acc[dt][e] *= alpha[e];

        // P C-layout -> A-layout via wave-private LDS slab (lockstep wave,
        // LDS ops are DScnt-ordered within the wave)
        asm volatile("s_wait_dscnt 0" ::: "memory");
        v16bf pF = load_a_frag(&plds[wave][0], 32, lane);

        // O += P (16x32) . V (32x128) from the staged V^T tile
#pragma unroll
        for (int dt = 0; dt < 8; ++dt) {
            v16bf vF = load_b_frag(&vbuf[buf][(dt * 16) * 32], 32, lane);
            acc[dt] = wmma_bf16(pF, vF, acc[dt]);
        }
        __syncthreads();                       // reads done before next overwrite
    }

#pragma unroll
    for (int dt = 0; dt < 8; ++dt)
#pragma unroll
        for (int e = 0; e < 8; ++e) {
            const int qrow = qbase + e + (grp << 3);
            float o = acc[dt][e] / lrow[e];
            O[(size_t)qrow * QDIM + h * HD + dt * 16 + col] = (__bf16)o;
        }
}

// ---------------------------------------------------------------------------
extern "C" void kernel_launch(void* const* d_in, const int* in_sizes, int n_in,
                              void* d_out, int out_size, void* d_ws, size_t ws_size,
                              hipStream_t stream) {
    const float* x  = (const float*)d_in[0];
    // d_in[1] = mask: causal, computed analytically in the kernel -> unused
    const float* wq = (const float*)d_in[2];
    const float* wk = (const float*)d_in[3];
    const float* wv = (const float*)d_in[4];
    const float* wo = (const float*)d_in[5];
    float* out = (float*)d_out;
    (void)in_sizes; (void)n_in; (void)out_size; (void)ws_size;

    char* ws = (char*)d_ws;
    size_t off = 0;
    auto take = [&](size_t bytes) -> char* {
        char* p = ws + off;
        off += (bytes + 255) & ~(size_t)255;
        return p;
    };
    __bf16* xb  = (__bf16*)take((size_t)L_SEQ * DMODEL * 2);
    __bf16* wqb = (__bf16*)take((size_t)QDIM  * DMODEL * 2);
    __bf16* wkb = (__bf16*)take((size_t)KVDIM * DMODEL * 2);
    __bf16* wvb = (__bf16*)take((size_t)KVDIM * DMODEL * 2);
    __bf16* wob = (__bf16*)take((size_t)DMODEL * QDIM  * 2);
    __bf16* Qb  = (__bf16*)take((size_t)L_SEQ * QDIM  * 2);
    __bf16* Kb  = (__bf16*)take((size_t)L_SEQ * KVDIM * 2);
    __bf16* Vb  = (__bf16*)take((size_t)L_SEQ * KVDIM * 2);
    __bf16* Vt  = (__bf16*)take((size_t)KVDIM * L_SEQ * 2);
    __bf16* Ab  = (__bf16*)take((size_t)L_SEQ * QDIM  * 2);
    // total workspace ~183 MB

    auto cvt = [&](const float* s, __bf16* d, size_t n) {
        int n4 = (int)(n / 4);
        cvt_f32_bf16x4<<<(n4 + 255) / 256, 256, 0, stream>>>(s, d, n4);
    };
    cvt(x,  xb,  (size_t)L_SEQ * DMODEL);
    cvt(wq, wqb, (size_t)QDIM  * DMODEL);
    cvt(wk, wkb, (size_t)KVDIM * DMODEL);
    cvt(wv, wvb, (size_t)KVDIM * DMODEL);
    cvt(wo, wob, (size_t)DMODEL * QDIM);

    // Projections (bf16 out); grid = (N/128, M/128)
    gemm_xwT<<<dim3(QDIM  / 128, L_SEQ / 128), 256, 0, stream>>>(xb, wqb, Qb, nullptr, L_SEQ, QDIM,  DMODEL);
    gemm_xwT<<<dim3(KVDIM / 128, L_SEQ / 128), 256, 0, stream>>>(xb, wkb, Kb, nullptr, L_SEQ, KVDIM, DMODEL);
    gemm_xwT<<<dim3(KVDIM / 128, L_SEQ / 128), 256, 0, stream>>>(xb, wvb, Vb, nullptr, L_SEQ, KVDIM, DMODEL);

    // RoPE (Q fused with SCALE = 1/sqrt(144) = 1/12), V transpose
    rope_inplace<<<(L_SEQ * NHEADS * 64) / 256, 256, 0, stream>>>(Qb, NHEADS, 1.0f / 12.0f);
    rope_inplace<<<(L_SEQ * NKV    * 64) / 256, 256, 0, stream>>>(Kb, NKV, 1.0f);
    transpose_bf16<<<(L_SEQ * KVDIM) / 256, 256, 0, stream>>>(Vb, Vt);

    // Attention
    flash_attn<<<dim3(L_SEQ / 128, NHEADS), 256, 0, stream>>>(Qb, Kb, Vt, Ab);

    // Output projection (fp32 out)
    gemm_xwT<<<dim3(DMODEL / 128, L_SEQ / 128), 256, 0, stream>>>(Ab, wob, nullptr, out, L_SEQ, DMODEL, QDIM);
}
